// MultiHeadAttention_52364241273282
// MI455X (gfx1250) — matrix operand order repaired and checked by reading, not compiled, not run on hardware
//
#include <hip/hip_runtime.h>

// ---------------------------------------------------------------------------
// Multi-head attention forward for MI455X (gfx1250), wave32 + WMMA bf16.
//   B=2, S=4096, DIM=512, HEADS=8, D_HEAD=64, SCALE=0.125
// Pipeline: fp32->bf16 convert, QKV WMMA GEMM (V stored transposed),
//           flash-attention with v_wmma_f32_16x16x32_bf16 + online softmax,
//           output-projection WMMA GEMM + bias (fp32 out).
// ---------------------------------------------------------------------------

typedef __attribute__((ext_vector_type(16))) __bf16 v16bf;
typedef __attribute__((ext_vector_type(8)))  __bf16 v8bf;
typedef __attribute__((ext_vector_type(4)))  __bf16 v4bf;
typedef __attribute__((ext_vector_type(8)))  float  v8f;

#define MHA_B      2
#define MHA_S      4096
#define MHA_DIM    512
#define MHA_H      8
#define MHA_DH     64
#define MHA_SCALE  0.125f
#define MHA_BS     (MHA_B * MHA_S)        // 8192

union FragU { v16bf v; v8bf h[2]; };

__device__ __forceinline__ v8f wmma_bf16(v16bf a, v16bf b, v8f c) {
  // emits v_wmma_f32_16x16x32_bf16
  return __builtin_amdgcn_wmma_f32_16x16x32_bf16(
      /*neg_a=*/false, a, /*neg_b=*/false, b,
      /*c_mod=*/(short)0, c, /*reuse_a=*/false, /*reuse_b=*/false);
}

// A-operand fragment: 16x32 bf16 tile from row-major [M,K] matrix.
// lane holds row (row0 + (lane&15)); K chunks [k0,k0+8) and [k0+16,k0+24),
// k0 = kbase + 8*(lane>>4).   (ISA 7.12.2, 16-bit A 16x32 layout)
__device__ __forceinline__ v16bf load_fragA(const __bf16* base, int ld,
                                            int row0, int kbase, int lane) {
  int r  = row0 + (lane & 15);
  int k0 = kbase + ((lane >> 4) << 3);
  const __bf16* p = base + (long)r * ld + k0;
  FragU u;
  u.h[0] = *reinterpret_cast<const v8bf*>(p);
  u.h[1] = *reinterpret_cast<const v8bf*>(p + 16);
  return u.v;
}

// B-operand fragment: 32x16 bf16 tile, sourced from row-major Bt[N,K]
// (i.e. B stored column-major). lane holds column (col0 + (lane&15));
// K chunks [k0,k0+8) and [k0+16,k0+24), k0 = kbase + 8*(lane>>4).
__device__ __forceinline__ v16bf load_fragB(const __bf16* base, int ld,
                                            int col0, int kbase, int lane) {
  int c  = col0 + (lane & 15);
  int k0 = kbase + ((lane >> 4) << 3);
  const __bf16* p = base + (long)c * ld + k0;
  FragU u;
  u.h[0] = *reinterpret_cast<const v8bf*>(p);
  u.h[1] = *reinterpret_cast<const v8bf*>(p + 16);
  return u.v;
}

// ---------------------------------------------------------------------------
// fp32 -> bf16 conversion, 4 elements per thread
// ---------------------------------------------------------------------------
__global__ void __launch_bounds__(256)
mha_cvt_bf16(const float* __restrict__ s, __bf16* __restrict__ d, int n) {
  int i = (blockIdx.x * blockDim.x + threadIdx.x) << 2;
  if (i < n) {
    float4 f = *reinterpret_cast<const float4*>(s + i);
    v4bf o;
    o.x = (__bf16)f.x; o.y = (__bf16)f.y; o.z = (__bf16)f.z; o.w = (__bf16)f.w;
    *reinterpret_cast<v4bf*>(d + i) = o;
  }
}

// ---------------------------------------------------------------------------
// QKV projection: C[bs,e] = sum_d xb[bs,d] * wqkv[e,d]   (M=8192,N=1536,K=512)
// Each wave computes a 16x64 output tile (4 accumulators, A-frag reuse).
// Epilogue scatters into Q (scaled), K [B,H,S,Dh], V transposed [B,H,Dh,S].
// ---------------------------------------------------------------------------
__global__ void __launch_bounds__(256)
mha_qkv_gemm(const __bf16* __restrict__ xb, const __bf16* __restrict__ wb,
             __bf16* __restrict__ qb, __bf16* __restrict__ kb,
             __bf16* __restrict__ vtb) {
  int lane = threadIdx.x & 31, wid = threadIdx.x >> 5;
  int gw  = blockIdx.x * 8 + wid;          // 0..12287
  int mt  = gw / 24;                       // 0..511  (M tile)
  int n64 = gw % 24;                       // 0..23   (64-wide N tile)
  int mrow = mt * 16;

  v8f c[4] = {};
  for (int kk = 0; kk < MHA_DIM; kk += 32) {
    __builtin_prefetch(xb + (long)mrow * MHA_DIM + kk + 64, 0, 0);
    v16bf af = load_fragA(xb, MHA_DIM, mrow, kk, lane);
#pragma unroll
    for (int t = 0; t < 4; ++t) {
      v16bf bf = load_fragB(wb, MHA_DIM, n64 * 64 + t * 16, kk, lane);
      c[t] = wmma_bf16(af, bf, c[t]);
    }
  }

  int n  = lane & 15;
  int mh = (lane >> 4) << 3;
#pragma unroll
  for (int t = 0; t < 4; ++t) {
    int e = n64 * 64 + t * 16 + n;         // 0..1535
    int section = e >> 9;                  // 0=Q 1=K 2=V
    int eh = e & 511;
    int h  = eh >> 6;
    int dd = eh & 63;
#pragma unroll
    for (int r = 0; r < 8; ++r) {
      int bs = mrow + mh + r;              // 0..8191
      int b  = bs >> 12, s = bs & 4095;
      int bh = b * MHA_H + h;
      float val = c[t][r];
      if (section == 0)
        qb[((long)(bh << 12) + s) * MHA_DH + dd] = (__bf16)(val * MHA_SCALE);
      else if (section == 1)
        kb[((long)(bh << 12) + s) * MHA_DH + dd] = (__bf16)val;
      else
        vtb[((long)(bh * MHA_DH + dd) << 12) + s] = (__bf16)val;
    }
  }
}

// ---------------------------------------------------------------------------
// Flash attention: one wave per 16-row Q tile; stream keys in 32-wide steps.
//   scores: 4 wmma / step ; softmax in fp32 ; P staged via LDS ; PV: 4 wmma.
// ---------------------------------------------------------------------------
__global__ void __launch_bounds__(256)
mha_flash_attn(const __bf16* __restrict__ qb, const __bf16* __restrict__ kbuf,
               const __bf16* __restrict__ vtb, __bf16* __restrict__ ao) {
  __shared__ __bf16 plds_all[8 * 16 * 32];
  int lane = threadIdx.x & 31, wid = threadIdx.x >> 5;
  __bf16* plds = plds_all + wid * (16 * 32);

  int gw = blockIdx.x * 8 + wid;           // 0..4095
  int bh = gw >> 8;                        // 0..15  (b*8+h)
  int qt = gw & 255;                       // q tile within (b,h)

  const __bf16* qp = qb   + (long)bh * MHA_S * MHA_DH;
  const __bf16* kp = kbuf + (long)bh * MHA_S * MHA_DH;
  const __bf16* vp = vtb  + (long)bh * MHA_DH * MHA_S;

  v16bf qf0 = load_fragA(qp, MHA_DH, qt * 16, 0, lane);
  v16bf qf1 = load_fragA(qp, MHA_DH, qt * 16, 32, lane);

  v8f o[4] = {};
  float mrow[8], lrow[8];
#pragma unroll
  for (int r = 0; r < 8; ++r) { mrow[r] = -1e30f; lrow[r] = 0.0f; }

  int n  = lane & 15;
  int mh = (lane >> 4) << 3;

  for (int kbase = 0; kbase < MHA_S; kbase += 32) {
    // ---- scores: two 16x16 tiles over [kbase, kbase+32) ----
    v8f s0 = {}, s1 = {};
    {
      v16bf b00 = load_fragB(kp, MHA_DH, kbase,      0,  lane);
      v16bf b01 = load_fragB(kp, MHA_DH, kbase,      32, lane);
      s0 = wmma_bf16(qf0, b00, s0);
      s0 = wmma_bf16(qf1, b01, s0);
      v16bf b10 = load_fragB(kp, MHA_DH, kbase + 16, 0,  lane);
      v16bf b11 = load_fragB(kp, MHA_DH, kbase + 16, 32, lane);
      s1 = wmma_bf16(qf0, b10, s1);
      s1 = wmma_bf16(qf1, b11, s1);
    }

    // ---- online softmax (rows split across lane halves per C layout) ----
#pragma unroll
    for (int r = 0; r < 8; ++r) {
      float tmax = fmaxf(s0[r], s1[r]);
      tmax = fmaxf(tmax, __shfl_xor(tmax, 1, 16));
      tmax = fmaxf(tmax, __shfl_xor(tmax, 2, 16));
      tmax = fmaxf(tmax, __shfl_xor(tmax, 4, 16));
      tmax = fmaxf(tmax, __shfl_xor(tmax, 8, 16));
      float mold = mrow[r];
      float mnew = fmaxf(mold, tmax);
      float corr = __expf(mold - mnew);
      float p0 = __expf(s0[r] - mnew);
      float p1 = __expf(s1[r] - mnew);
      float rs = p0 + p1;
      rs += __shfl_xor(rs, 1, 16);
      rs += __shfl_xor(rs, 2, 16);
      rs += __shfl_xor(rs, 4, 16);
      rs += __shfl_xor(rs, 8, 16);
      mrow[r] = mnew;
      lrow[r] = lrow[r] * corr + rs;
      o[0][r] *= corr; o[1][r] *= corr; o[2][r] *= corr; o[3][r] *= corr;
      int m = mh + r;                       // C-layout row
      plds[m * 32 + n]      = (__bf16)p0;   // cols kbase..+15
      plds[m * 32 + 16 + n] = (__bf16)p1;   // cols kbase+16..+31
    }
    // same-wave DS ordering guarantees store->load visibility

    // ---- P (16x32, A layout) @ V (32x64 via Vt rows) ----
    v16bf pf = load_fragA(plds, 32, 0, 0, lane);
#pragma unroll
    for (int t = 0; t < 4; ++t) {
      v16bf vf = load_fragB(vp, MHA_S, t * 16, kbase, lane);
      o[t] = wmma_bf16(pf, vf, o[t]);
    }
  }

  // ---- normalize & store to [B,S,DIM] bf16 (heads re-interleaved) ----
  int b = bh >> 3, h = bh & 7;
#pragma unroll
  for (int r = 0; r < 8; ++r) {
    float inv = 1.0f / lrow[r];
    int m    = mh + r;
    int srow = qt * 16 + m;
#pragma unroll
    for (int t = 0; t < 4; ++t) {
      int col = h * MHA_DH + t * 16 + n;
      ao[((long)((b << 12) + srow) << 9) + col] = (__bf16)(o[t][r] * inv);
    }
  }
}

// ---------------------------------------------------------------------------
// Output projection: out[bs,e] = ao[bs,:] . w_out[e,:] + b_out[e]  (fp32 out)
// ---------------------------------------------------------------------------
__global__ void __launch_bounds__(256)
mha_oproj_gemm(const __bf16* __restrict__ ao, const __bf16* __restrict__ wob,
               const float* __restrict__ bias, float* __restrict__ out) {
  int lane = threadIdx.x & 31, wid = threadIdx.x >> 5;
  int gw  = blockIdx.x * 8 + wid;          // 0..4095
  int mt  = gw >> 3;                       // 0..511
  int n64 = gw & 7;                        // 0..7
  int mrow = mt * 16;

  v8f c[4] = {};
  for (int kk = 0; kk < MHA_DIM; kk += 32) {
    __builtin_prefetch(ao + (long)mrow * MHA_DIM + kk + 64, 0, 0);
    v16bf af = load_fragA(ao, MHA_DIM, mrow, kk, lane);
#pragma unroll
    for (int t = 0; t < 4; ++t) {
      v16bf bf = load_fragB(wob, MHA_DIM, n64 * 64 + t * 16, kk, lane);
      c[t] = wmma_bf16(af, bf, c[t]);
    }
  }

  int n  = lane & 15;
  int mh = (lane >> 4) << 3;
#pragma unroll
  for (int t = 0; t < 4; ++t) {
    int e = n64 * 64 + t * 16 + n;
    float bv = bias[e];
#pragma unroll
    for (int r = 0; r < 8; ++r) {
      int bs = mrow + mh + r;
      out[(long)bs * MHA_DIM + e] = c[t][r] + bv;
    }
  }
}

// ---------------------------------------------------------------------------
// Host launcher
// ---------------------------------------------------------------------------
extern "C" void kernel_launch(void* const* d_in, const int* in_sizes, int n_in,
                              void* d_out, int out_size, void* d_ws, size_t ws_size,
                              hipStream_t stream) {
  const float* x     = (const float*)d_in[0];   // [2,4096,512]
  const float* w_qkv = (const float*)d_in[1];   // [1536,512]
  const float* w_out = (const float*)d_in[2];   // [512,512]
  const float* b_out = (const float*)d_in[3];   // [512]
  float* out = (float*)d_out;                   // [2,4096,512] fp32

  char* ws = (char*)d_ws;
  // workspace layout (bytes)
  __bf16* xb    = (__bf16*)(ws);                       //  8 MB  [8192,512]
  __bf16* wqkvb = (__bf16*)(ws + 8388608);             //  1.5MB [1536,512]
  __bf16* woutb = (__bf16*)(ws + 9961472);             //  0.5MB [512,512]
  __bf16* qb    = (__bf16*)(ws + 10485760);            //  8 MB  [B,H,S,Dh] (pre-scaled)
  __bf16* kb    = (__bf16*)(ws + 18874368);            //  8 MB  [B,H,S,Dh]
  __bf16* vtb   = (__bf16*)(ws + 27262976);            //  8 MB  [B,H,Dh,S]
  __bf16* aob   = (__bf16*)(ws + 35651584);            //  8 MB  [B,S,DIM]

  // fp32 -> bf16 converts
  mha_cvt_bf16<<<(MHA_BS * MHA_DIM) / 4 / 256, 256, 0, stream>>>(x, xb, MHA_BS * MHA_DIM);
  mha_cvt_bf16<<<(3 * MHA_DIM * MHA_DIM) / 4 / 256, 256, 0, stream>>>(w_qkv, wqkvb, 3 * MHA_DIM * MHA_DIM);
  mha_cvt_bf16<<<(MHA_DIM * MHA_DIM) / 4 / 256, 256, 0, stream>>>(w_out, woutb, MHA_DIM * MHA_DIM);

  // QKV projection: 512 M-tiles * 24 N64-tiles = 12288 waves / 8 per block
  mha_qkv_gemm<<<1536, 256, 0, stream>>>(xb, wqkvb, qb, kb, vtb);

  // attention: 4096 q-tile waves / 8 per block
  mha_flash_attn<<<512, 256, 0, stream>>>(qb, kb, vtb, aob);

  // output projection: 512*8 = 4096 waves / 8 per block
  mha_oproj_gemm<<<512, 256, 0, stream>>>(aob, woutb, b_out, out);
}
